// CAM_SE_Module_71330816852397
// MI455X (gfx1250) — matrix operand order repaired and checked by reading, not compiled, not run on hardware
//
#include <hip/hip_runtime.h>
#include <hip/hip_bf16.h>
#include <math.h>

// ---------------- problem constants ----------------
#define BATCH 16
#define C_DIM 256
#define NPIX  9216      // 96*96
#define RDIM  32        // C/8
#define KSPLIT 4
#define KLEN  (NPIX / KSPLIT)   // 2304, divisible by 64

// LDS tile: 64 rows x 64 k, padded to 72 elements/row (144 B rows: 16B-aligned,
// 36-bank stride -> conflict-free 16-lane column access)
#define LDS_STRIDE 72

typedef __attribute__((ext_vector_type(16))) __bf16 bf16x16;
typedef __attribute__((ext_vector_type(8)))  __bf16 bf16x8;
typedef __attribute__((ext_vector_type(8)))  float  f32x8;

// ---------------- helpers ----------------
__device__ inline void split_bf16(float f, __bf16& hi, __bf16& lo) {
    hi = (__bf16)f;               // RNE fptrunc
    lo = (__bf16)(f - (float)hi); // residual
}

// A-operand fragment (16x32 bf16, contiguous k order used here, same as load_fragB):
// lane<16 : M=lane, K = kb..kb+15 contiguous in v[0..15]
// lane>=16: same M, K offset +16.
__device__ inline bf16x16 load_fragA(const __bf16* s, int row, int kbase) {
    const bf16x8 a = *(const bf16x8*)(s + row * LDS_STRIDE + kbase);
    const bf16x8 b = *(const bf16x8*)(s + row * LDS_STRIDE + kbase + 8);
    bf16x16 r;
#pragma unroll
    for (int i = 0; i < 8; ++i) { r[i] = a[i]; r[i + 8] = b[i]; }
    return r;
}

// B-operand fragment (32x16 bf16): lane<16 holds N=lane with K=0..15 contiguous,
// lane>=16 holds K=16..31. LDS row = output column, K contiguous in the row.
__device__ inline bf16x16 load_fragB(const __bf16* s, int row, int kbase) {
    const bf16x8 a = *(const bf16x8*)(s + row * LDS_STRIDE + kbase);
    const bf16x8 b = *(const bf16x8*)(s + row * LDS_STRIDE + kbase + 8);
    bf16x16 r;
#pragma unroll
    for (int i = 0; i < 8; ++i) { r[i] = a[i]; r[i + 8] = b[i]; }
    return r;
}

__device__ inline f32x8 wmma_bf16(bf16x16 a, bf16x16 b, f32x8 c) {
    return __builtin_amdgcn_wmma_f32_16x16x32_bf16(
        false, a, false, b, (short)0, c, false, false);
}

// Async global->LDS copy, 16B per lane (ISA 10.1/15.18.3, ASYNCcnt-tracked).
__device__ inline void async_copy_b128(unsigned int lds_byte_off, const void* gptr) {
    asm volatile("global_load_async_to_lds_b128 %0, %1, off"
                 :: "v"(lds_byte_off), "v"((unsigned long long)(uintptr_t)gptr)
                 : "memory");
}

// ---------------- kernel 1: global average pool ----------------
__global__ __launch_bounds__(256) void pool_kernel(const float* __restrict__ x,
                                                   float* __restrict__ pooled) {
    const int bc = blockIdx.x;                     // (b*C + c)
    const float* p = x + (size_t)bc * NPIX;
    float s = 0.f;
    for (int i = threadIdx.x; i < NPIX; i += 256) s += p[i];
    __shared__ float red[256];
    red[threadIdx.x] = s;
    __syncthreads();
    for (int off = 128; off > 0; off >>= 1) {
        if (threadIdx.x < off) red[threadIdx.x] += red[threadIdx.x + off];
        __syncthreads();
    }
    if (threadIdx.x == 0) pooled[bc] = red[0] * (1.0f / NPIX);
}

// ---------------- kernel 2: SE MLP (tiny, one block) ----------------
__global__ __launch_bounds__(256) void se_kernel(const float* __restrict__ pooled,
                                                 const float* __restrict__ w1,
                                                 const float* __restrict__ b1,
                                                 const float* __restrict__ w2,
                                                 const float* __restrict__ b2,
                                                 float* __restrict__ gate) {
    __shared__ float h[RDIM];
    const int tid = threadIdx.x;
    for (int b = 0; b < BATCH; ++b) {
        if (tid < RDIM) {
            float s = b1[tid];
            for (int k = 0; k < C_DIM; ++k) s += pooled[b * C_DIM + k] * w1[tid * C_DIM + k];
            h[tid] = fmaxf(s, 0.f);
        }
        __syncthreads();
        float s = b2[tid];
#pragma unroll
        for (int k = 0; k < RDIM; ++k) s += w2[tid * RDIM + k] * h[k];
        gate[b * C_DIM + tid] = 1.0f / (1.0f + __expf(-s));
        __syncthreads();
    }
}

// ---------------- kernel 3: energy = V*V^T (split-K, 3xBF16 emulated fp32) ----
// grid: 16(b) * 4(mt) * 4(nt) * 4(ks) = 1024 blocks, 512 threads (16 waves 4x4)
// Software pipeline: next chunk staged in VGPRs while WMMAs run on current tile.
__global__ __launch_bounds__(512) void energy_kernel(const float* __restrict__ x,
                                                     float* __restrict__ part) {
    const int bx = blockIdx.x;
    const int ks = bx & 3;
    const int nt = (bx >> 2) & 3;
    const int mt = (bx >> 4) & 3;
    const int b  = bx >> 6;
    const int m0 = mt * 64, n0 = nt * 64, k0 = ks * KLEN;

    __shared__ alignas(16) __bf16 sAhi[64 * LDS_STRIDE];
    __shared__ alignas(16) __bf16 sAlo[64 * LDS_STRIDE];
    __shared__ alignas(16) __bf16 sBhi[64 * LDS_STRIDE];
    __shared__ alignas(16) __bf16 sBlo[64 * LDS_STRIDE];

    const float* vb = x + (size_t)b * C_DIM * NPIX;
    const int tid = threadIdx.x, lane = tid & 31, wid = tid >> 5;
    const int wm = wid >> 2, wn = wid & 3;
    const int lrow = tid >> 4;          // 0..31 (rows lrow and lrow+32)
    const int lcol = (tid & 15) * 4;    // float4 column

    float4 ra0, ra1, rb0, rb1;
    auto loadG = [&](int kk) {
        ra0 = *(const float4*)(vb + (size_t)(m0 + lrow) * NPIX + kk + lcol);
        ra1 = *(const float4*)(vb + (size_t)(m0 + lrow + 32) * NPIX + kk + lcol);
        rb0 = *(const float4*)(vb + (size_t)(n0 + lrow) * NPIX + kk + lcol);
        rb1 = *(const float4*)(vb + (size_t)(n0 + lrow + 32) * NPIX + kk + lcol);
    };
    auto storeS = [&]() {
#pragma unroll
        for (int i = 0; i < 4; ++i) {
            __bf16 h, l;
            split_bf16((&ra0.x)[i], h, l);
            sAhi[lrow * LDS_STRIDE + lcol + i] = h;
            sAlo[lrow * LDS_STRIDE + lcol + i] = l;
            split_bf16((&ra1.x)[i], h, l);
            sAhi[(lrow + 32) * LDS_STRIDE + lcol + i] = h;
            sAlo[(lrow + 32) * LDS_STRIDE + lcol + i] = l;
            split_bf16((&rb0.x)[i], h, l);
            sBhi[lrow * LDS_STRIDE + lcol + i] = h;
            sBlo[lrow * LDS_STRIDE + lcol + i] = l;
            split_bf16((&rb1.x)[i], h, l);
            sBhi[(lrow + 32) * LDS_STRIDE + lcol + i] = h;
            sBlo[(lrow + 32) * LDS_STRIDE + lcol + i] = l;
        }
    };

    f32x8 acc = {};
    loadG(k0);
    for (int kk = k0; kk < k0 + KLEN; kk += 64) {
        __syncthreads();                 // previous tile fully consumed
        storeS();
        if (kk + 64 < k0 + KLEN) loadG(kk + 64);   // in flight during WMMAs
        __syncthreads();                 // tile visible to all waves
#pragma unroll
        for (int st = 0; st < 2; ++st) {
            const int ka = st * 32 + ((lane < 16) ? 0 : 16);
            const int kb = st * 32 + ((lane < 16) ? 0 : 16);
            const int ar = wm * 16 + (lane & 15);
            const int br = wn * 16 + (lane & 15);
            const bf16x16 Ah = load_fragA(sAhi, ar, ka);
            const bf16x16 Al = load_fragA(sAlo, ar, ka);
            const bf16x16 Bh = load_fragB(sBhi, br, kb);
            const bf16x16 Bl = load_fragB(sBlo, br, kb);
            acc = wmma_bf16(Ah, Bh, acc);   // Hi*Hi
            acc = wmma_bf16(Ah, Bl, acc);   // Hi*Lo
            acc = wmma_bf16(Al, Bh, acc);   // Lo*Hi  (~tf32-accurate energy)
        }
    }

    // C/D layout: vgpr j -> row j (+8 for lanes>=16), col = lane%16
    float* dst = part + ((size_t)ks * BATCH + b) * C_DIM * C_DIM;
    const int col   = n0 + wn * 16 + (lane & 15);
    const int rbase = m0 + wm * 16 + ((lane >> 4) << 3);
#pragma unroll
    for (int j = 0; j < 8; ++j)
        dst[(size_t)(rbase + j) * C_DIM + col] = acc[j];
}

// ---------------- kernel 4: row softmax of (rowmax - e) == softmax(-e) -------
// wave per row; sums KSPLIT partials deterministically; emits bf16 attention.
__global__ __launch_bounds__(256) void softmax_kernel(const float* __restrict__ part,
                                                      __bf16* __restrict__ att) {
    const int row  = blockIdx.x * 8 + (threadIdx.x >> 5);   // 0..4095
    const int lane = threadIdx.x & 31;
    const int b = row >> 8, c = row & 255;
    const size_t KS = (size_t)BATCH * C_DIM * C_DIM;
    const float* p0 = part + ((size_t)b * C_DIM + c) * C_DIM;

    float e[8];
#pragma unroll
    for (int i = 0; i < 8; ++i) {
        const int col = lane + i * 32;
        e[i] = p0[col] + p0[KS + col] + p0[2 * KS + col] + p0[3 * KS + col];
    }
    float mn = e[0];
#pragma unroll
    for (int i = 1; i < 8; ++i) mn = fminf(mn, e[i]);
    for (int off = 16; off > 0; off >>= 1) mn = fminf(mn, __shfl_xor(mn, off, 32));

    float p[8], sum = 0.f;
#pragma unroll
    for (int i = 0; i < 8; ++i) { p[i] = __expf(mn - e[i]); sum += p[i]; }
    for (int off = 16; off > 0; off >>= 1) sum += __shfl_xor(sum, off, 32);
    const float inv = 1.0f / sum;

    __bf16* dst = att + ((size_t)b * C_DIM + c) * C_DIM;
#pragma unroll
    for (int i = 0; i < 8; ++i) dst[lane + i * 32] = (__bf16)(p[i] * inv);
}

// ---------------- kernel 5: out = gamma * gate * (att @ V) + x ---------------
// grid: 16(b) * 4(mt) * 144(nt) = 9216 blocks, 512 threads (16 waves 4x4).
// Attention tile: bf16, copied global->LDS with GLOBAL_LOAD_ASYNC_TO_LDS_B128
// into a double buffer (ASYNCcnt). V tile: register-staged, transposed+converted
// on LDS store.
__global__ __launch_bounds__(512) void attn_out_kernel(const float* __restrict__ x,
                                                       const __bf16* __restrict__ att,
                                                       const float* __restrict__ gate,
                                                       const float* __restrict__ gamma,
                                                       float* __restrict__ out) {
    const int bx = blockIdx.x;
    const int nt = bx % 144;
    const int mt = (bx / 144) & 3;
    const int b  = bx / (144 * 4);
    const int n0 = nt * 64, m0 = mt * 64;

    __shared__ alignas(16) __bf16 sA[2][64 * LDS_STRIDE];  // attention (double buf)
    __shared__ alignas(16) __bf16 sBT[64 * LDS_STRIDE];    // V tile transposed [n][k]

    const __bf16* attb = att + (size_t)b * C_DIM * C_DIM;
    const float*  xb   = x   + (size_t)b * C_DIM * NPIX;
    const int tid = threadIdx.x, lane = tid & 31, wid = tid >> 5;
    const int wm = wid >> 2, wn = wid & 3;
    // A async-copy coords: 8 threads/row, 16B (8 bf16) each -> 64x64 in one shot
    const int arow = tid >> 3, acol = (tid & 7) * 8;
    // V staging coords
    const int lrow = tid >> 4;          // rows lrow, lrow+32
    const int lcol = (tid & 15) * 4;

    auto issueA = [&](int buf, int kk) {
        const unsigned int lds =
            (unsigned int)(uintptr_t)&sA[buf][arow * LDS_STRIDE + acol];
        async_copy_b128(lds, attb + (size_t)(m0 + arow) * C_DIM + kk + acol);
    };
    float4 rv0, rv1;
    auto loadV = [&](int kk) {
        rv0 = *(const float4*)(xb + (size_t)(kk + lrow) * NPIX + n0 + lcol);
        rv1 = *(const float4*)(xb + (size_t)(kk + lrow + 32) * NPIX + n0 + lcol);
    };
    auto storeV = [&]() {
#pragma unroll
        for (int i = 0; i < 4; ++i) {   // transpose-on-store: LDS row = column n
            sBT[(lcol + i) * LDS_STRIDE + lrow]      = (__bf16)((&rv0.x)[i]);
            sBT[(lcol + i) * LDS_STRIDE + lrow + 32] = (__bf16)((&rv1.x)[i]);
        }
    };

    f32x8 acc = {};
    issueA(0, 0);
    loadV(0);
    int cur = 0;
    for (int kk = 0; kk < C_DIM; kk += 64, cur ^= 1) {
        __syncthreads();                 // previous tile fully consumed
        storeV();
        const bool nxt = (kk + 64) < C_DIM;
        if (nxt) {                       // overlap next chunk with WMMAs
            issueA(cur ^ 1, kk + 64);
            loadV(kk + 64);
            asm volatile("s_wait_asynccnt 0x1" ::: "memory"); // current A done
        } else {
            asm volatile("s_wait_asynccnt 0x0" ::: "memory");
        }
        __syncthreads();                 // LDS visible to all waves
#pragma unroll
        for (int st = 0; st < 2; ++st) {
            const bf16x16 A = load_fragA(sA[cur], wm * 16 + (lane & 15),
                                         st * 32 + ((lane < 16) ? 0 : 16));
            const bf16x16 B = load_fragB(sBT, wn * 16 + (lane & 15),
                                         st * 32 + ((lane < 16) ? 0 : 16));
            acc = wmma_bf16(A, B, acc);
        }
    }

    const float g = gamma[0];
    const int col   = n0 + wn * 16 + (lane & 15);
    const int rbase = m0 + wm * 16 + ((lane >> 4) << 3);
#pragma unroll
    for (int j = 0; j < 8; ++j) {
        const int c = rbase + j;
        const float gt = gate[b * C_DIM + c];
        const size_t idx = ((size_t)b * C_DIM + c) * NPIX + col;
        out[idx] = g * acc[j] * gt + x[idx];
    }
}

// ---------------- launcher ----------------
extern "C" void kernel_launch(void* const* d_in, const int* in_sizes, int n_in,
                              void* d_out, int out_size, void* d_ws, size_t ws_size,
                              hipStream_t stream) {
    const float* x     = (const float*)d_in[0];
    const float* gamma = (const float*)d_in[1];
    const float* w1    = (const float*)d_in[2];
    const float* b1    = (const float*)d_in[3];
    const float* w2    = (const float*)d_in[4];
    const float* b2    = (const float*)d_in[5];
    float* out = (float*)d_out;

    // workspace layout (fully rewritten every call -> deterministic):
    //   part   : KSPLIT * B * C * C f32 = 16 MB
    //   att    : B * C * C bf16         =  2 MB
    //   pooled : B * C f32              = 16 KB
    //   gate   : B * C f32              = 16 KB
    float*  part   = (float*)d_ws;
    __bf16* att    = (__bf16*)(part + (size_t)KSPLIT * BATCH * C_DIM * C_DIM);
    float*  pooled = (float*)(att + (size_t)BATCH * C_DIM * C_DIM);
    float*  gate   = pooled + BATCH * C_DIM;

    pool_kernel<<<BATCH * C_DIM, 256, 0, stream>>>(x, pooled);
    se_kernel<<<1, 256, 0, stream>>>(pooled, w1, b1, w2, b2, gate);
    energy_kernel<<<BATCH * 4 * 4 * KSPLIT, 512, 0, stream>>>(x, part);
    softmax_kernel<<<(BATCH * C_DIM) / 8, 256, 0, stream>>>(part, att);
    attn_out_kernel<<<BATCH * 4 * 144, 512, 0, stream>>>(x, att, gate, gamma, out);
}
